// SinGVP_67886253080586
// MI455X (gfx1250) — matrix operand order repaired and checked by reading, not compiled, not run on hardware
//
#include <hip/hip_runtime.h>
#include <hip/hip_bf16.h>
#include <math.h>

#define NN 20000
#define NE 320000
#define NG 32
#define EPSV 1e-8f

typedef __attribute__((ext_vector_type(16))) _Float16 v16h;
typedef __attribute__((ext_vector_type(8)))  float    v8f;
typedef _Float16 h16;

struct alignas(16) F4 { float a, b, c, d; };
union V16HU { v16h v; F4 q[2]; h16 h[16]; };

__device__ __forceinline__ float sigmf(float x) { return 1.f / (1.f + __expf(-x)); }

// ---------------------------------------------------------------------------
// 16xN row-block GEMM: D[16][N] = A(16 x 32*KSTEPS, f16 in LDS) * Bsw + bias
// Bsw is pre-swizzled f16 so each lane's fragment is 16 contiguous halves.
// mode 0: write f16 (+relu) into ldsH (stride okpad), all padded n
// mode 1: atomicAdd f32 into gOut[rows[m]*rowStride + n], n < nValid
// mode 2: store f32 to gOut[(rowBase+m)*rowStride + n], n < nValid
// mode 3: store f32 into ldsF (stride okpad), n < nValid
// ---------------------------------------------------------------------------
template <int KSTEPS, int NTILES>
__device__ __forceinline__ void gemm16(
    const h16* __restrict__ A, int kpad,
    const h16* __restrict__ Bsw, const float* __restrict__ bias,
    bool relu, int mode,
    h16* ldsH, float* ldsF, int okpad,
    float* gOut, const int* rows, int rowBase, int rowStride, int nValid,
    int lane)
{
  const int m0 = lane & 15, hi = lane >> 4;
  const h16* arow = A + m0 * kpad;
  for (int nt = 0; nt < NTILES; ++nt) {
    const int n = nt * 16 + m0;
    const float bv = (n < nValid) ? bias[n] : 0.f;
    v8f c;
#pragma unroll
    for (int r = 0; r < 8; ++r) c[r] = bv;
#pragma unroll
    for (int ks = 0; ks < KSTEPS; ++ks) {
      V16HU a, b;
      const h16* ap = arow + ks * 32 + hi * 16;     // K groups {0-15} / {16-31}
      a.q[0] = *reinterpret_cast<const F4*>(ap);
      a.q[1] = *reinterpret_cast<const F4*>(ap + 8);
      const h16* bp = Bsw + ((size_t)(ks * NTILES + nt) * 32 + lane) * 16;
      b.q[0] = *reinterpret_cast<const F4*>(bp);
      b.q[1] = *reinterpret_cast<const F4*>(bp + 8);
      c = __builtin_amdgcn_wmma_f32_16x16x32_f16(false, a.v, false, b.v,
                                                 (short)0, c, false, false);
    }
#pragma unroll
    for (int r = 0; r < 8; ++r) {
      const int mr = r + 8 * hi;                    // D row per ISA C/D layout
      float val = c[r];
      if (relu) val = fmaxf(val, 0.f);
      if (mode == 0) {
        ldsH[mr * okpad + n] = (h16)val;
      } else if (mode == 3) {
        if (n < nValid) ldsF[mr * okpad + n] = val;
      } else if (n < nValid) {
        const int row = rows ? rows[mr] : (rowBase + mr);
        if (mode == 1) atomicAdd(gOut + (size_t)row * rowStride + n, val);
        else           gOut[(size_t)row * rowStride + n] = val;
      }
    }
  }
}

// ---------------------------------------------------------------------------
// weight swizzle: W[K][N] f32 -> f16 fragments laid out so pack_B is 2x16B
// ---------------------------------------------------------------------------
__global__ void k_swz(const float* __restrict__ W, h16* __restrict__ out,
                      int K, int N, int NT, int total)
{
  int i = blockIdx.x * blockDim.x + threadIdx.x;
  if (i >= total) return;
  int j  = i & 15;
  int L  = (i >> 4) & 31;
  int g  = i >> 9;            // ks*NT + nt
  int nt = g % NT;
  int ks = g / NT;
  int k   = ks * 32 + (L >> 4) * 16 + j;
  int col = nt * 16 + (L & 15);
  out[i] = (k < K && col < N) ? (h16)W[(size_t)k * N + col] : (h16)0.f;
}

__global__ void k_zero(float* p, int n)
{
  int i = blockIdx.x * blockDim.x + threadIdx.x;
  if (i < n) p[i] = 0.f;
}

__global__ void k_cnt(const int* __restrict__ eidx, float* __restrict__ cnt)
{
  int e = blockIdx.x * blockDim.x + threadIdx.x;
  if (e < NE) atomicAdd(cnt + eidx[NE + e], 1.f);
}
__global__ void k_gcnt(const int* __restrict__ batch, float* __restrict__ gcnt)
{
  int n = blockIdx.x * blockDim.x + threadIdx.x;
  if (n < NN) atomicAdd(gcnt + batch[n], 1.f);
}

// ---------------------------------------------------------------------------
// node embed: LN + GVP(6s,3v -> 100s,16v), no activations
// ---------------------------------------------------------------------------
__global__ void k_node_embed(const float* __restrict__ hVs, const float* __restrict__ hVv,
                             const float* __restrict__ lnG, const float* __restrict__ lnB,
                             const float* __restrict__ wh, const float* __restrict__ wsw,
                             const float* __restrict__ wsb, const float* __restrict__ wv,
                             float* __restrict__ sCur, h16* __restrict__ sHalf,
                             float* __restrict__ vCur)
{
  int n = blockIdx.x * blockDim.x + threadIdx.x;
  if (n >= NN) return;
  float s[6], vl[9], vh[48];
  float mu = 0.f;
  for (int c = 0; c < 6; ++c) { s[c] = hVs[(size_t)n * 6 + c]; mu += s[c]; }
  mu /= 6.f;
  float var = 0.f;
  for (int c = 0; c < 6; ++c) { float d = s[c] - mu; var += d * d; }
  float rstd = rsqrtf(var / 6.f + 1e-5f);
  for (int c = 0; c < 6; ++c) s[c] = (s[c] - mu) * rstd * lnG[c] + lnB[c];
  float sv = 0.f;
  for (int k = 0; k < 3; ++k) {
    float s2 = 0.f;
    for (int d = 0; d < 3; ++d) { float x = hVv[(size_t)n * 9 + k * 3 + d]; vl[k * 3 + d] = x; s2 += x * x; }
    sv += fmaxf(s2, EPSV);
  }
  float rn = rsqrtf(sv / 3.f);
  for (int c = 0; c < 9; ++c) vl[c] *= rn;
  for (int h = 0; h < 16; ++h)
    for (int d = 0; d < 3; ++d) {
      float a = 0.f;
      for (int k = 0; k < 3; ++k) a += vl[k * 3 + d] * wh[k * 16 + h];
      vh[d * 16 + h] = a;
    }
  float sin22[22];
  for (int c = 0; c < 6; ++c) sin22[c] = s[c];
  for (int h = 0; h < 16; ++h) {
    float s2 = 0.f;
    for (int d = 0; d < 3; ++d) { float x = vh[d * 16 + h]; s2 += x * x; }
    sin22[6 + h] = sqrtf(fmaxf(s2, EPSV));
  }
  for (int o = 0; o < 100; ++o) {
    float a = wsb[o];
    for (int c = 0; c < 22; ++c) a += sin22[c] * wsw[c * 100 + o];
    sCur[(size_t)n * 100 + o] = a;
    sHalf[(size_t)n * 100 + o] = (h16)a;
  }
  for (int o = 0; o < 16; ++o)
    for (int d = 0; d < 3; ++d) {
      float a = 0.f;
      for (int h = 0; h < 16; ++h) a += vh[d * 16 + h] * wv[h * 16 + o];
      vCur[(size_t)n * 48 + o * 3 + d] = a;
    }
}

// ---------------------------------------------------------------------------
// edge embed: LN + GVP(32s,1v -> 32s,1v), no activations
// ---------------------------------------------------------------------------
__global__ void k_edge_embed(const float* __restrict__ hEs, const float* __restrict__ hEv,
                             const float* __restrict__ lnG, const float* __restrict__ lnB,
                             const float* __restrict__ wh, const float* __restrict__ wsw,
                             const float* __restrict__ wsb, const float* __restrict__ wv,
                             h16* __restrict__ esH, float* __restrict__ evF)
{
  int e = blockIdx.x * blockDim.x + threadIdx.x;
  if (e >= NE) return;
  float s[32];
  float mu = 0.f;
  for (int c = 0; c < 32; ++c) { s[c] = hEs[(size_t)e * 32 + c]; mu += s[c]; }
  mu /= 32.f;
  float var = 0.f;
  for (int c = 0; c < 32; ++c) { float d = s[c] - mu; var += d * d; }
  float rstd = rsqrtf(var / 32.f + 1e-5f);
  for (int c = 0; c < 32; ++c) s[c] = (s[c] - mu) * rstd * lnG[c] + lnB[c];
  float v0 = hEv[(size_t)e * 3 + 0], v1 = hEv[(size_t)e * 3 + 1], v2 = hEv[(size_t)e * 3 + 2];
  float rn = rsqrtf(fmaxf(v0 * v0 + v1 * v1 + v2 * v2, EPSV));  // NV=1: mean == value
  v0 *= rn; v1 *= rn; v2 *= rn;
  float w00 = wh[0];
  float h0 = v0 * w00, h1 = v1 * w00, h2 = v2 * w00;
  float vn = sqrtf(fmaxf(h0 * h0 + h1 * h1 + h2 * h2, EPSV));
  for (int o = 0; o < 32; ++o) {
    float a = wsb[o];
    for (int c = 0; c < 32; ++c) a += s[c] * wsw[c * 32 + o];
    a += vn * wsw[32 * 32 + o];
    esH[(size_t)e * 32 + o] = (h16)a;
  }
  float wv00 = wv[0];
  evF[(size_t)e * 3 + 0] = h0 * wv00;
  evF[(size_t)e * 3 + 1] = h1 * wv00;
  evF[(size_t)e * 3 + 2] = h2 * wv00;
}

// ---------------------------------------------------------------------------
// edge message kernel: one wave = 16 edges, 4 chained GVPs, WMMA for ws GEMMs
// ---------------------------------------------------------------------------
struct MsgW {
  const float* wh[4];
  const float* wv[4];
  const float* bias[4];
  const h16*   sw[4];
};

__global__ void __launch_bounds__(32)
k_msg(const float* __restrict__ vCur, const h16* __restrict__ sHalf,
      const h16* __restrict__ esH, const float* __restrict__ evF,
      const int* __restrict__ eidx,
      float* __restrict__ dhs, float* __restrict__ dhv, MsgW W)
{
  __shared__ __align__(16) h16   A0[16 * 288];
  __shared__ __align__(16) h16   A1[16 * 128];
  __shared__ __align__(16) float MV[16 * 99];
  __shared__ __align__(16) float VH[16 * 99];
  __shared__ int DSTI[16];
  const int lane = threadIdx.x;
  const int e0 = blockIdx.x * 16;
  const int* src = eidx;
  const int* dst = eidx + NE;

  if (lane < 16) DSTI[lane] = dst[e0 + lane];
  // gather scalars: [s_src(100) | es(32) | s_dst(100)] as f16
  for (int idx = lane; idx < 16 * 232; idx += 32) {
    int m = idx / 232, c = idx - 232 * m, e = e0 + m;
    h16 v;
    if (c < 100)      v = sHalf[(size_t)src[e] * 100 + c];
    else if (c < 132) v = esH[(size_t)e * 32 + (c - 100)];
    else              v = sHalf[(size_t)dst[e] * 100 + (c - 132)];
    A0[m * 288 + c] = v;
  }
  for (int idx = lane; idx < 16 * 56; idx += 32) { int m = idx / 56, c = idx - 56 * m; A0[m * 288 + 232 + c] = (h16)0.f; }
  for (int idx = lane; idx < 16 * 12; idx += 32) { int m = idx / 12, c = idx - 12 * m; A1[m * 128 + 116 + c] = (h16)0.f; }
  // gather vectors: [v_src(16) | ev(1) | v_dst(16)] x 3
  for (int idx = lane; idx < 16 * 99; idx += 32) {
    int m = idx / 99, r = idx - 99 * m, k = r / 3, d = r - 3 * k, e = e0 + m;
    float v;
    if (k < 16)       v = vCur[(size_t)src[e] * 48 + k * 3 + d];
    else if (k == 16) v = evF[(size_t)e * 3 + d];
    else              v = vCur[(size_t)dst[e] * 48 + (k - 17) * 3 + d];
    MV[m * 99 + k * 3 + d] = v;
  }
  __syncthreads();

  // ---- stage 0: SI=232 VI=33 H=33 VO=16, relu + sigmoid-gate
  for (int idx = lane; idx < 16 * 99; idx += 32) {
    int m = idx / 99, r = idx - 99 * m, d = r / 33, h = r - 33 * d;
    float acc = 0.f;
    for (int k = 0; k < 33; ++k) acc += MV[m * 99 + k * 3 + d] * W.wh[0][k * 33 + h];
    VH[m * 99 + d * 33 + h] = acc;
  }
  __syncthreads();
  for (int idx = lane; idx < 16 * 33; idx += 32) {
    int m = idx / 33, h = idx - 33 * m;
    float s = 0.f;
    for (int d = 0; d < 3; ++d) { float x = VH[m * 99 + d * 33 + h]; s += x * x; }
    A0[m * 288 + 232 + h] = (h16)sqrtf(fmaxf(s, EPSV));
  }
  __syncthreads();
  gemm16<9, 7>(A0, 288, W.sw[0], W.bias[0], true, 0, A1, nullptr, 128,
               nullptr, nullptr, 0, 0, 100, lane);
  __syncthreads();
  for (int idx = lane; idx < 16 * 16; idx += 32) {
    int m = idx / 16, o = idx - 16 * m;
    float a0 = 0.f, a1 = 0.f, a2 = 0.f;
    for (int h = 0; h < 33; ++h) {
      float w = W.wv[0][h * 16 + o];
      a0 += VH[m * 99 + h] * w;
      a1 += VH[m * 99 + 33 + h] * w;
      a2 += VH[m * 99 + 66 + h] * w;
    }
    float g = sigmf(sqrtf(fmaxf(a0 * a0 + a1 * a1 + a2 * a2, EPSV)));
    MV[m * 48 + o * 3 + 0] = a0 * g;
    MV[m * 48 + o * 3 + 1] = a1 * g;
    MV[m * 48 + o * 3 + 2] = a2 * g;
  }
  __syncthreads();

  // ---- stages 1..3: SI=100 VI=16 H=16 VO=16 (stage 3 no act, scatter)
  for (int st = 1; st <= 3; ++st) {
    h16* Ain  = (st & 1) ? A1 : A0;   // 128-wide view
    h16* Aout = (st & 1) ? A0 : A1;
    const bool fin = (st == 3);
    for (int idx = lane; idx < 16 * 48; idx += 32) {
      int m = idx / 48, r = idx - 48 * m, d = r / 16, h = r - 16 * d;
      float acc = 0.f;
      for (int k = 0; k < 16; ++k) acc += MV[m * 48 + k * 3 + d] * W.wh[st][k * 16 + h];
      VH[m * 48 + d * 16 + h] = acc;
    }
    __syncthreads();
    for (int idx = lane; idx < 16 * 16; idx += 32) {
      int m = idx / 16, h = idx - 16 * m;
      float s = 0.f;
      for (int d = 0; d < 3; ++d) { float x = VH[m * 48 + d * 16 + h]; s += x * x; }
      Ain[m * 128 + 100 + h] = (h16)sqrtf(fmaxf(s, EPSV));
    }
    if (st == 1)
      for (int idx = lane; idx < 16 * 12; idx += 32) { int m = idx / 12, c = idx - 12 * m; A0[m * 128 + 116 + c] = (h16)0.f; }
    __syncthreads();
    if (!fin)
      gemm16<4, 7>(Ain, 128, W.sw[st], W.bias[st], true, 0, Aout, nullptr, 128,
                   nullptr, nullptr, 0, 0, 100, lane);
    else
      gemm16<4, 7>(Ain, 128, W.sw[st], W.bias[st], false, 1, nullptr, nullptr, 0,
                   dhs, DSTI, 0, 100, 100, lane);
    __syncthreads();
    for (int idx = lane; idx < 16 * 16; idx += 32) {
      int m = idx / 16, o = idx - 16 * m;
      float a0 = 0.f, a1 = 0.f, a2 = 0.f;
      for (int h = 0; h < 16; ++h) {
        float w = W.wv[st][h * 16 + o];
        a0 += VH[m * 48 + h] * w;
        a1 += VH[m * 48 + 16 + h] * w;
        a2 += VH[m * 48 + 32 + h] * w;
      }
      if (!fin) {
        float g = sigmf(sqrtf(fmaxf(a0 * a0 + a1 * a1 + a2 * a2, EPSV)));
        MV[m * 48 + o * 3 + 0] = a0 * g;
        MV[m * 48 + o * 3 + 1] = a1 * g;
        MV[m * 48 + o * 3 + 2] = a2 * g;
      } else {
        int dn = DSTI[m];
        atomicAdd(dhv + (size_t)dn * 48 + o * 3 + 0, a0);
        atomicAdd(dhv + (size_t)dn * 48 + o * 3 + 1, a1);
        atomicAdd(dhv + (size_t)dn * 48 + o * 3 + 2, a2);
      }
    }
    __syncthreads();
  }
}

// ---------------------------------------------------------------------------
// node update: residual + LN0 + ff-GVP (WMMA) + residual + LN1, store outs
// ---------------------------------------------------------------------------
struct NodeW {
  const float *ffwh, *ffwv, *ffb;
  const h16* ffsw;
  const float *g0, *b0, *g1, *b1;
};

__global__ void __launch_bounds__(32)
k_node_update(float* __restrict__ sCur, h16* __restrict__ sHalf, float* __restrict__ vCur,
              const float* __restrict__ dhs, const float* __restrict__ dhv,
              const float* __restrict__ cnt,
              float* __restrict__ outS, float* __restrict__ outV, NodeW W)
{
  __shared__ __align__(16) float S1[16 * 100];
  __shared__ __align__(16) float V1[16 * 48];
  __shared__ __align__(16) h16   A[16 * 128];
  __shared__ __align__(16) float SFF[16 * 112];
  __shared__ __align__(16) float VH[16 * 48];
  __shared__ __align__(16) float V2[16 * 48];
  const int lane = threadIdx.x;
  const int base = blockIdx.x * 16;

  for (int idx = lane; idx < 16 * 100; idx += 32) {
    int m = idx / 100, c = idx - 100 * m, n = base + m;
    float cn = fmaxf(cnt[n], 1.f);
    S1[idx] = sCur[(size_t)n * 100 + c] + dhs[(size_t)n * 100 + c] / cn;
  }
  for (int idx = lane; idx < 16 * 48; idx += 32) {
    int m = idx / 48, c = idx - 48 * m, n = base + m;
    float cn = fmaxf(cnt[n], 1.f);
    V1[idx] = vCur[(size_t)n * 48 + c] + dhv[(size_t)n * 48 + c] / cn;
  }
  __syncthreads();
  if (lane < 16) {
    float mu = 0.f;
    for (int c = 0; c < 100; ++c) mu += S1[lane * 100 + c];
    mu /= 100.f;
    float var = 0.f;
    for (int c = 0; c < 100; ++c) { float d = S1[lane * 100 + c] - mu; var += d * d; }
    float rstd = rsqrtf(var / 100.f + 1e-5f);
    for (int c = 0; c < 100; ++c) S1[lane * 100 + c] = (S1[lane * 100 + c] - mu) * rstd * W.g0[c] + W.b0[c];
    float sv = 0.f;
    for (int k = 0; k < 16; ++k) {
      float s2 = 0.f;
      for (int d = 0; d < 3; ++d) { float x = V1[lane * 48 + k * 3 + d]; s2 += x * x; }
      sv += fmaxf(s2, EPSV);
    }
    float rn = rsqrtf(sv / 16.f);
    for (int c = 0; c < 48; ++c) V1[lane * 48 + c] *= rn;
  }
  __syncthreads();
  for (int idx = lane; idx < 16 * 100; idx += 32) { int m = idx / 100, c = idx - 100 * m; A[m * 128 + c] = (h16)S1[idx]; }
  for (int idx = lane; idx < 16 * 48; idx += 32) {
    int m = idx / 48, r = idx - 48 * m, d = r / 16, h = r - 16 * d;
    float acc = 0.f;
    for (int k = 0; k < 16; ++k) acc += V1[m * 48 + k * 3 + d] * W.ffwh[k * 16 + h];
    VH[m * 48 + d * 16 + h] = acc;
  }
  __syncthreads();
  for (int idx = lane; idx < 16 * 16; idx += 32) {
    int m = idx / 16, h = idx - 16 * m;
    float s = 0.f;
    for (int d = 0; d < 3; ++d) { float x = VH[m * 48 + d * 16 + h]; s += x * x; }
    A[m * 128 + 100 + h] = (h16)sqrtf(fmaxf(s, EPSV));
  }
  for (int idx = lane; idx < 16 * 12; idx += 32) { int m = idx / 12, c = idx - 12 * m; A[m * 128 + 116 + c] = (h16)0.f; }
  __syncthreads();
  gemm16<4, 7>(A, 128, W.ffsw, W.ffb, false, 3, nullptr, SFF, 112,
               nullptr, nullptr, 0, 0, 100, lane);
  __syncthreads();
  for (int idx = lane; idx < 16 * 16; idx += 32) {
    int m = idx / 16, o = idx - 16 * m;
    float a0 = 0.f, a1 = 0.f, a2 = 0.f;
    for (int h = 0; h < 16; ++h) {
      float w = W.ffwv[h * 16 + o];
      a0 += VH[m * 48 + h] * w;
      a1 += VH[m * 48 + 16 + h] * w;
      a2 += VH[m * 48 + 32 + h] * w;
    }
    V2[m * 48 + o * 3 + 0] = a0;
    V2[m * 48 + o * 3 + 1] = a1;
    V2[m * 48 + o * 3 + 2] = a2;
  }
  __syncthreads();
  if (lane < 16) {
    const int n = base + lane;
    float mu = 0.f;
    for (int c = 0; c < 100; ++c) { float x = S1[lane * 100 + c] + SFF[lane * 112 + c]; SFF[lane * 112 + c] = x; mu += x; }
    mu /= 100.f;
    float var = 0.f;
    for (int c = 0; c < 100; ++c) { float d = SFF[lane * 112 + c] - mu; var += d * d; }
    float rstd = rsqrtf(var / 100.f + 1e-5f);
    for (int c = 0; c < 100; ++c) {
      float y = (SFF[lane * 112 + c] - mu) * rstd * W.g1[c] + W.b1[c];
      sCur[(size_t)n * 100 + c] = y;
      sHalf[(size_t)n * 100 + c] = (h16)y;
      outS[(size_t)n * 100 + c] = y;
    }
    float sv = 0.f;
    for (int k = 0; k < 16; ++k) {
      float s2 = 0.f;
      for (int d = 0; d < 3; ++d) { float x = V1[lane * 48 + k * 3 + d] + V2[lane * 48 + k * 3 + d]; s2 += x * x; }
      sv += fmaxf(s2, EPSV);
    }
    float rn = rsqrtf(sv / 16.f);
    for (int c = 0; c < 48; ++c) {
      float y = (V1[lane * 48 + c] + V2[lane * 48 + c]) * rn;
      vCur[(size_t)n * 48 + c] = y;
      outV[(size_t)n * 48 + c] = y;
    }
  }
}

// ---------------------------------------------------------------------------
// Wout: LN over concat(300s, 48v) + GVP(348 -> 300, relu), WMMA
// ---------------------------------------------------------------------------
__global__ void __launch_bounds__(32)
k_wout(const float* __restrict__ outS, const float* __restrict__ outV,
       const float* __restrict__ lnG, const float* __restrict__ lnB,
       const float* __restrict__ wh, const h16* __restrict__ sw,
       const float* __restrict__ bias, float* __restrict__ outNodes)
{
  __shared__ __align__(16) h16   A[16 * 352];
  __shared__ __align__(16) float CV[16 * 144];
  __shared__ __align__(16) float VH[16 * 144];
  const int lane = threadIdx.x;
  const int base = blockIdx.x * 16;
  if (lane < 16) {
    const int n = base + lane;
    float mu = 0.f;
    for (int l = 0; l < 3; ++l)
      for (int c = 0; c < 100; ++c) mu += outS[((size_t)l * NN + n) * 100 + c];
    mu /= 300.f;
    float var = 0.f;
    for (int l = 0; l < 3; ++l)
      for (int c = 0; c < 100; ++c) { float d = outS[((size_t)l * NN + n) * 100 + c] - mu; var += d * d; }
    float rstd = rsqrtf(var / 300.f + 1e-5f);
    for (int l = 0; l < 3; ++l)
      for (int c = 0; c < 100; ++c) {
        int j = l * 100 + c;
        A[lane * 352 + j] = (h16)((outS[((size_t)l * NN + n) * 100 + c] - mu) * rstd * lnG[j] + lnB[j]);
      }
    float sv = 0.f;
    for (int l = 0; l < 3; ++l)
      for (int k = 0; k < 16; ++k) {
        float s2 = 0.f;
        for (int d = 0; d < 3; ++d) { float x = outV[((size_t)l * NN + n) * 48 + k * 3 + d]; s2 += x * x; }
        sv += fmaxf(s2, EPSV);
      }
    float rn = rsqrtf(sv / 48.f);
    for (int l = 0; l < 3; ++l)
      for (int k = 0; k < 16; ++k)
        for (int d = 0; d < 3; ++d)
          CV[lane * 144 + (l * 16 + k) * 3 + d] = outV[((size_t)l * NN + n) * 48 + k * 3 + d] * rn;
  }
  __syncthreads();
  for (int idx = lane; idx < 16 * 144; idx += 32) {
    int m = idx / 144, r = idx - 144 * m, d = r / 48, h = r - 48 * d;
    float acc = 0.f;
    for (int k = 0; k < 48; ++k) acc += CV[m * 144 + k * 3 + d] * wh[k * 48 + h];
    VH[m * 144 + d * 48 + h] = acc;
  }
  __syncthreads();
  for (int idx = lane; idx < 16 * 48; idx += 32) {
    int m = idx / 48, h = idx - 48 * m;
    float s = 0.f;
    for (int d = 0; d < 3; ++d) { float x = VH[m * 144 + d * 48 + h]; s += x * x; }
    A[m * 352 + 300 + h] = (h16)sqrtf(fmaxf(s, EPSV));
  }
  for (int idx = lane; idx < 16 * 4; idx += 32) { int m = idx / 4, c = idx - 4 * m; A[m * 352 + 348 + c] = (h16)0.f; }
  __syncthreads();
  gemm16<11, 19>(A, 352, sw, bias, true, 2, nullptr, nullptr, 0,
                 outNodes, nullptr, base, 300, 300, lane);
}

__global__ void k_scatter(const float* __restrict__ outNodes, const int* __restrict__ batch,
                          float* __restrict__ out)
{
  int i = blockIdx.x * blockDim.x + threadIdx.x;
  if (i >= NN * 300) return;
  int n = i / 300, c = i - 300 * n;
  atomicAdd(out + (size_t)batch[n] * 300 + c, outNodes[i]);
}
__global__ void k_div(float* __restrict__ out, const float* __restrict__ gcnt)
{
  int i = blockIdx.x * blockDim.x + threadIdx.x;
  if (i < NG * 300) out[i] /= fmaxf(gcnt[i / 300], 1.f);
}

// ---------------------------------------------------------------------------
extern "C" void kernel_launch(void* const* d_in, const int* in_sizes, int n_in,
                              void* d_out, int out_size, void* d_ws, size_t ws_size,
                              hipStream_t stream)
{
  const float* hVs = (const float*)d_in[0];
  const float* hVv = (const float*)d_in[1];
  const float* hEs = (const float*)d_in[2];
  const float* hEv = (const float*)d_in[3];
  const float* P[89];
  for (int i = 0; i < 89; ++i) P[i] = (const float*)d_in[4 + i];
  const int* eidx  = (const int*)d_in[93];
  const int* batch = (const int*)d_in[94];
  float* out = (float*)d_out;
  // param leaf order (jax tree: dict keys sorted):
  // 0 We_gvp.wh 1 We_gvp.ws.b 2 We_gvp.ws.w 3 We_gvp.wv 4 We_ln.b 5 We_ln.g
  // 6 Wout_gvp.wh 7 Wout_gvp.ws.b 8 Wout_gvp.ws.w 9 Wout_ln.b 10 Wout_ln.g
  // 11 Wv_gvp.wh 12 Wv_gvp.ws.b 13 Wv_gvp.ws.w 14 Wv_gvp.wv 15 Wv_ln.b 16 Wv_ln.g
  // layer l (base 17+24l): ff(wh,wsb,wsw,wv) msg0..3(wh,wsb,wsw,wv) n0.b n0.g n1.b n1.g

  char* ws = (char*)d_ws;
  size_t off = 0;
  auto carve = [&](size_t bytes) -> char* {
    char* p = ws + off;
    off = (off + bytes + 255) & ~(size_t)255;
    return p;
  };
  h16* swMsg0[3]; h16* swMsg[3][3]; h16* swFF[3];
  for (int l = 0; l < 3; ++l) {
    swMsg0[l] = (h16*)carve((size_t)288 * 112 * 2);
    for (int j = 0; j < 3; ++j) swMsg[l][j] = (h16*)carve((size_t)128 * 112 * 2);
    swFF[l] = (h16*)carve((size_t)128 * 112 * 2);
  }
  h16*   swWout  = (h16*)carve((size_t)352 * 304 * 2);
  float* sCur    = (float*)carve((size_t)NN * 100 * 4);
  h16*   sHalf   = (h16*)carve((size_t)NN * 100 * 2);
  float* vCur    = (float*)carve((size_t)NN * 48 * 4);
  h16*   esH     = (h16*)carve((size_t)NE * 32 * 2);
  float* evF     = (float*)carve((size_t)NE * 3 * 4);
  float* dhs     = (float*)carve((size_t)NN * 100 * 4);
  float* dhv     = (float*)carve((size_t)NN * 48 * 4);
  float* cnt     = (float*)carve((size_t)NN * 4);
  float* outS    = (float*)carve((size_t)3 * NN * 100 * 4);
  float* outV    = (float*)carve((size_t)3 * NN * 48 * 4);
  float* outNode = (float*)carve((size_t)NN * 300 * 4);
  float* gcnt    = (float*)carve(32 * 4);

  auto swz = [&](const float* W, h16* dst, int K, int N, int KP, int NP) {
    int NT = NP / 16, total = KP * NP;
    k_swz<<<(total + 255) / 256, 256, 0, stream>>>(W, dst, K, N, NT, total);
  };
  for (int l = 0; l < 3; ++l) {
    int b = 17 + 24 * l;
    swz(P[b + 6],  swMsg0[l],   265, 100, 288, 112);
    swz(P[b + 10], swMsg[l][0], 116, 100, 128, 112);
    swz(P[b + 14], swMsg[l][1], 116, 100, 128, 112);
    swz(P[b + 18], swMsg[l][2], 116, 100, 128, 112);
    swz(P[b + 2],  swFF[l],     116, 100, 128, 112);
  }
  swz(P[8], swWout, 348, 300, 352, 304);

  k_zero<<<(NN + 255) / 256, 256, 0, stream>>>(cnt, NN);
  k_zero<<<1, 32, 0, stream>>>(gcnt, 32);
  k_zero<<<(NG * 300 + 255) / 256, 256, 0, stream>>>(out, NG * 300);
  k_cnt<<<(NE + 255) / 256, 256, 0, stream>>>(eidx, cnt);
  k_gcnt<<<(NN + 255) / 256, 256, 0, stream>>>(batch, gcnt);

  k_node_embed<<<(NN + 127) / 128, 128, 0, stream>>>(hVs, hVv, P[16], P[15], P[11], P[13], P[12], P[14],
                                                     sCur, sHalf, vCur);
  k_edge_embed<<<(NE + 127) / 128, 128, 0, stream>>>(hEs, hEv, P[5], P[4], P[0], P[2], P[1], P[3],
                                                     esH, evF);

  for (int l = 0; l < 3; ++l) {
    int b = 17 + 24 * l;
    k_zero<<<(NN * 100 + 255) / 256, 256, 0, stream>>>(dhs, NN * 100);
    k_zero<<<(NN * 48 + 255) / 256, 256, 0, stream>>>(dhv, NN * 48);
    MsgW W;
    W.wh[0] = P[b + 4];  W.bias[0] = P[b + 5];  W.wv[0] = P[b + 7];  W.sw[0] = swMsg0[l];
    W.wh[1] = P[b + 8];  W.bias[1] = P[b + 9];  W.wv[1] = P[b + 11]; W.sw[1] = swMsg[l][0];
    W.wh[2] = P[b + 12]; W.bias[2] = P[b + 13]; W.wv[2] = P[b + 15]; W.sw[2] = swMsg[l][1];
    W.wh[3] = P[b + 16]; W.bias[3] = P[b + 17]; W.wv[3] = P[b + 19]; W.sw[3] = swMsg[l][2];
    k_msg<<<NE / 16, 32, 0, stream>>>(vCur, sHalf, esH, evF, eidx, dhs, dhv, W);
    NodeW NW;
    NW.ffwh = P[b + 0]; NW.ffb = P[b + 1]; NW.ffsw = swFF[l]; NW.ffwv = P[b + 3];
    NW.b0 = P[b + 20]; NW.g0 = P[b + 21]; NW.b1 = P[b + 22]; NW.g1 = P[b + 23];
    k_node_update<<<NN / 16, 32, 0, stream>>>(sCur, sHalf, vCur, dhs, dhv, cnt,
                                              outS + (size_t)l * NN * 100,
                                              outV + (size_t)l * NN * 48, NW);
  }
  k_wout<<<NN / 16, 32, 0, stream>>>(outS, outV, P[10], P[9], P[6], swWout, P[7], outNode);
  k_scatter<<<(NN * 300 + 255) / 256, 256, 0, stream>>>(outNode, batch, out);
  k_div<<<(NG * 300 + 255) / 256, 256, 0, stream>>>(out, gcnt);
}
